// Attention_87428354277736
// MI455X (gfx1250) — matrix operand order repaired and checked by reading, not compiled, not run on hardware
//
#include <hip/hip_runtime.h>
#include <hip/hip_bf16.h>
#include <math.h>

// ---------------------------------------------------------------------------
// Clustered attention for MI455X (gfx1250), bf16 WMMA path.
//   q = (x||clusters) @ Wq * scale ; kv = x @ Wkv
//   x branch   : flash attention (softmax over keys), fused, LDS P-transpose
//   cluster br : column softmax over cluster axis + row renorm, WMMA PV
//   outputs    : (a @ v) @ Wo  for both branches
// All matmuls: v_wmma_f32_16x16x32_bf16 (head_dim==32 == WMMA K depth).
// token_sizes log-bias is per-query constant under softmax over k -> no-op.
// ---------------------------------------------------------------------------

typedef unsigned short u16_t;
typedef __attribute__((ext_vector_type(16))) __bf16 v16bf;
typedef __attribute__((ext_vector_type(8)))  float  v8f;

#define K_HEADS 16
#define K_HD    32
#define K_TOK   2048
#define K_CLU   256
#define K_NQ    2304      // K_TOK + K_CLU
#define K_D     512
#define K_B     2

static __device__ __forceinline__ u16_t f32_to_bf16(float f) {
  unsigned int u = __float_as_uint(f);
  u += 0x7FFFu + ((u >> 16) & 1u);        // round-to-nearest-even
  return (u16_t)(u >> 16);
}

// A-fragment, 16x32 bf16, A row-major [M][K] (lda in elements).
// lane<16 : M=lane, K = {0..7, 16..23}; lane>=16 : M=lane-16, K = {8..15, 24..31}
static __device__ __forceinline__ v16bf load_frag_a(const u16_t* A, int lda) {
  const int lane = threadIdx.x & 31;
  const int m    = lane & 15;
  const int kh   = (lane >> 4) << 3;       // 0 or 8
  const u16_t* p = A + m * lda + kh;
  union { uint4 q[2]; v16bf v; } u;
  u.q[0] = *(const uint4*)(p);             // K = kh .. kh+7
  u.q[1] = *(const uint4*)(p + 16);        // K = kh+16 .. kh+23
  return u.v;
}

// B-fragment, 32x16 bf16, from transposed storage Bt row-major [N][K].
// lane<16 : N=lane, K = {0..7, 16..23}; lane>=16 : N=lane-16, K = {8..15, 24..31}
static __device__ __forceinline__ v16bf load_frag_b(const u16_t* Bt, int ldb) {
  const int lane = threadIdx.x & 31;
  const int n    = lane & 15;
  const int kh   = (lane >> 4) << 3;       // 0 or 8
  const u16_t* p = Bt + n * ldb + kh;
  union { uint4 q[2]; v16bf v; } u;
  u.q[0] = *(const uint4*)(p);
  u.q[1] = *(const uint4*)(p + 16);
  return u.v;
}

static __device__ __forceinline__ v8f wmma_bf16(v16bf a, v16bf b, v8f c) {
  return __builtin_amdgcn_wmma_f32_16x16x32_bf16(
      /*neg_a=*/false, a, /*neg_b=*/false, b,
      /*c_mod=*/(short)0, c, /*reuse_a=*/false, /*reuse_b=*/false);
}

// ---------------------------------------------------------------------------
// Prep kernels
// ---------------------------------------------------------------------------

// Wt[n*K + k] = bf16(W[k*N + n])   (W is [K][N] f32)
__global__ void transpose_to_bf16(const float* __restrict__ W,
                                  u16_t* __restrict__ Wt, int K, int N) {
  int i = blockIdx.x * blockDim.x + threadIdx.x;
  if (i >= K * N) return;
  int n = i / K, k = i % K;
  Wt[(size_t)n * K + k] = f32_to_bf16(W[(size_t)k * N + n]);
}

// Xbf[b][0..2047][:] = x, Xbf[b][2048..2303][:] = clusters  (bf16)
__global__ void pack_inputs(const float* __restrict__ x,
                            const float* __restrict__ clusters,
                            u16_t* __restrict__ Xbf) {
  size_t i = (size_t)blockIdx.x * blockDim.x + threadIdx.x;
  if (i >= (size_t)K_B * K_NQ * K_D) return;
  int c = (int)(i % K_D);
  int r = (int)((i / K_D) % K_NQ);
  int b = (int)(i / ((size_t)K_NQ * K_D));
  float v = (r < K_TOK) ? x[((size_t)b * K_TOK + r) * K_D + c]
                        : clusters[((size_t)b * K_CLU + (r - K_TOK)) * K_D + c];
  Xbf[i] = f32_to_bf16(v);
}

// ---------------------------------------------------------------------------
// Projection GEMMs (wave = 16x16 output tile, K-loop step 32)
// ---------------------------------------------------------------------------

// Q = (Xbf @ Wq) * scale, written head-split [b][h][m][hd] bf16
__global__ void qproj_kernel(const u16_t* __restrict__ Xbf,
                             const u16_t* __restrict__ WqT,
                             u16_t* __restrict__ Q) {
  int wid = (blockIdx.x * blockDim.x + threadIdx.x) >> 5;
  const int TPB = (K_NQ / 16) * (K_D / 16);              // 144*32
  if (wid >= K_B * TPB) return;
  int b = wid / TPB, t = wid % TPB;
  int mt = t / (K_D / 16), nt = t % (K_D / 16);
  const u16_t* A  = Xbf + ((size_t)b * K_NQ + mt * 16) * K_D;
  const u16_t* Bt = WqT + (size_t)nt * 16 * K_D;
  v8f acc = {};
  for (int k = 0; k < K_D; k += 32) {
    if (k + 64 <= K_D) { __builtin_prefetch(A + k + 32, 0, 1); __builtin_prefetch(Bt + k + 32, 0, 1); }
    acc = wmma_bf16(load_frag_a(A + k, K_D), load_frag_b(Bt + k, K_D), acc);
  }
  const int lane = threadIdx.x & 31, half = lane >> 4;
  const int col = nt * 16 + (lane & 15);
  const int h = col >> 5, hd = col & 31;
  const float scale = 0.17677669529663688f;              // 32^-0.5
  u16_t* dst = Q + ((size_t)(b * K_HEADS + h) * K_NQ) * K_HD + hd;
  #pragma unroll
  for (int r = 0; r < 8; ++r) {
    int m = mt * 16 + r + 8 * half;
    dst[(size_t)m * K_HD] = f32_to_bf16(acc[r] * scale);
  }
}

// kv = x @ Wkv;  n<512 -> Kbuf[b][h][m][hd] ;  n>=512 -> Vt[b][h][hd][m]
__global__ void kvproj_kernel(const u16_t* __restrict__ Xbf,
                              const u16_t* __restrict__ WkvT,
                              u16_t* __restrict__ Kbuf,
                              u16_t* __restrict__ Vt) {
  int wid = (blockIdx.x * blockDim.x + threadIdx.x) >> 5;
  const int TPB = (K_TOK / 16) * (2 * K_D / 16);         // 128*64
  if (wid >= K_B * TPB) return;
  int b = wid / TPB, t = wid % TPB;
  int mt = t / (2 * K_D / 16), nt = t % (2 * K_D / 16);
  const u16_t* A  = Xbf + ((size_t)b * K_NQ + mt * 16) * K_D;  // x rows only
  const u16_t* Bt = WkvT + (size_t)nt * 16 * K_D;
  v8f acc = {};
  for (int k = 0; k < K_D; k += 32) {
    if (k + 64 <= K_D) { __builtin_prefetch(A + k + 32, 0, 1); __builtin_prefetch(Bt + k + 32, 0, 1); }
    acc = wmma_bf16(load_frag_a(A + k, K_D), load_frag_b(Bt + k, K_D), acc);
  }
  const int lane = threadIdx.x & 31, half = lane >> 4;
  const int col = nt * 16 + (lane & 15);
  if (col < K_D) {                                       // K part
    const int h = col >> 5, hd = col & 31;
    u16_t* dst = Kbuf + ((size_t)(b * K_HEADS + h) * K_TOK) * K_HD + hd;
    #pragma unroll
    for (int r = 0; r < 8; ++r) {
      int m = mt * 16 + r + 8 * half;
      dst[(size_t)m * K_HD] = f32_to_bf16(acc[r]);
    }
  } else {                                               // V part, transposed
    const int cc = col - K_D, h = cc >> 5, hd = cc & 31;
    u16_t* dst = Vt + ((size_t)(b * K_HEADS + h) * K_HD + hd) * K_TOK;
    #pragma unroll
    for (int r = 0; r < 8; ++r) {
      int m = mt * 16 + r + 8 * half;
      dst[m] = f32_to_bf16(acc[r]);
    }
  }
}

// ---------------------------------------------------------------------------
// x branch: fused flash attention per (b,h), 64 queries/block, 4 waves
// ---------------------------------------------------------------------------
__global__ void __launch_bounds__(128)
flash_x_kernel(const u16_t* __restrict__ Q, const u16_t* __restrict__ Kbuf,
               const u16_t* __restrict__ Vt, u16_t* __restrict__ AO) {
  const int bh   = blockIdx.x >> 5;                      // 0..31 (b*16+h)
  const int qt   = blockIdx.x & 31;                      // 64-query tile
  const int w    = threadIdx.x >> 5;                     // wave 0..3
  const int lane = threadIdx.x & 31;
  const int half = lane >> 4;
  const int qbase = qt * 64 + w * 16;

  const u16_t* Qp = Q    + ((size_t)bh * K_NQ + qbase) * K_HD;
  const u16_t* Kp = Kbuf + (size_t)bh * K_TOK * K_HD;
  const u16_t* Vp = Vt   + (size_t)bh * K_HD * K_TOK;

  __shared__ __align__(16) u16_t ldsP[4][16 * 32];       // per-wave P tile
  u16_t* pbuf = &ldsP[w][0];

  v16bf aq = load_frag_a(Qp, K_HD);                      // 16 queries x hd32
  v8f o0 = {}, o1 = {};
  float mrun[8], lrun[8];
  #pragma unroll
  for (int r = 0; r < 8; ++r) { mrun[r] = -INFINITY; lrun[r] = 0.f; }

  for (int kt = 0; kt < K_TOK; kt += 32) {
    v16bf bk0 = load_frag_b(Kp + (size_t)(kt +  0) * K_HD, K_HD);
    v16bf bk1 = load_frag_b(Kp + (size_t)(kt + 16) * K_HD, K_HD);
    v8f zero = {};
    v8f s0 = wmma_bf16(aq, bk0, zero);                   // keys kt..kt+15
    v8f s1 = wmma_bf16(aq, bk1, zero);                   // keys kt+16..kt+31

    #pragma unroll
    for (int r = 0; r < 8; ++r) {
      float pm = fmaxf(s0[r], s1[r]);                    // row-max across cols
      pm = fmaxf(pm, __shfl_xor(pm, 1, 32));
      pm = fmaxf(pm, __shfl_xor(pm, 2, 32));
      pm = fmaxf(pm, __shfl_xor(pm, 4, 32));
      pm = fmaxf(pm, __shfl_xor(pm, 8, 32));
      float mn = fmaxf(mrun[r], pm);
      float al = __expf(mrun[r] - mn);
      float e0 = __expf(s0[r] - mn);
      float e1 = __expf(s1[r] - mn);
      float ps = e0 + e1;
      ps += __shfl_xor(ps, 1, 32);
      ps += __shfl_xor(ps, 2, 32);
      ps += __shfl_xor(ps, 4, 32);
      ps += __shfl_xor(ps, 8, 32);
      lrun[r] = lrun[r] * al + ps;
      mrun[r] = mn;
      o0[r] *= al;
      o1[r] *= al;
      int row = r + 8 * half;                            // C-layout -> LDS [16][32]
      pbuf[row * 32 + (lane & 15)]      = f32_to_bf16(e0);
      pbuf[row * 32 + 16 + (lane & 15)] = f32_to_bf16(e1);
    }

    v16bf ap  = load_frag_a(pbuf, 32);                   // wave-local, ds-ordered
    v16bf bv0 = load_frag_b(Vp + kt, K_TOK);             // hd 0..15
    v16bf bv1 = load_frag_b(Vp + (size_t)16 * K_TOK + kt, K_TOK); // hd 16..31
    o0 = wmma_bf16(ap, bv0, o0);
    o1 = wmma_bf16(ap, bv1, o1);
  }

  const int b = bh >> 4, h = bh & 15;
  #pragma unroll
  for (int r = 0; r < 8; ++r) {
    float inv = 1.f / lrun[r];
    int m = qbase + r + 8 * half;
    u16_t* d0 = AO + ((size_t)b * K_NQ + m) * K_D + h * K_HD;
    d0[(lane & 15)]      = f32_to_bf16(o0[r] * inv);
    d0[16 + (lane & 15)] = f32_to_bf16(o1[r] * inv);
  }
}

// ---------------------------------------------------------------------------
// Cluster branch
// ---------------------------------------------------------------------------

// Sc[bh][c][k] = q_c . k   (single WMMA per 16x16 tile; K = head_dim = 32)
__global__ void cscore_kernel(const u16_t* __restrict__ Q,
                              const u16_t* __restrict__ Kbuf,
                              float* __restrict__ Sc) {
  int wid = (blockIdx.x * blockDim.x + threadIdx.x) >> 5;
  if (wid >= K_B * K_HEADS * (K_CLU / 16) * (K_TOK / 16)) return;
  int bh = wid >> 11;                                    // 16*128 tiles per bh
  int t  = wid & 2047;
  int mt = t >> 7, nt = t & 127;
  const u16_t* A  = Q    + ((size_t)bh * K_NQ + K_TOK + mt * 16) * K_HD;
  const u16_t* Bt = Kbuf + ((size_t)bh * K_TOK + nt * 16) * K_HD;
  v8f acc = {};
  acc = wmma_bf16(load_frag_a(A, K_HD), load_frag_b(Bt, K_HD), acc);
  const int lane = threadIdx.x & 31, half = lane >> 4;
  float* dst = Sc + (size_t)bh * K_CLU * K_TOK;
  #pragma unroll
  for (int r = 0; r < 8; ++r) {
    int row = mt * 16 + r + 8 * half;
    int col = nt * 16 + (lane & 15);
    dst[(size_t)row * K_TOK + col] = acc[r];
  }
}

// softmax over cluster axis (columns of Sc), in place
__global__ void csoftmax_col_kernel(float* __restrict__ Sc) {
  int col = blockIdx.x * blockDim.x + threadIdx.x;       // 65536 columns
  if (col >= K_B * K_HEADS * K_TOK) return;
  float* p = Sc + (size_t)(col >> 11) * K_CLU * K_TOK + (col & (K_TOK - 1));
  float mx = -INFINITY;
  for (int i = 0; i < K_CLU; ++i) mx = fmaxf(mx, p[(size_t)i * K_TOK]);
  float s = 0.f;
  for (int i = 0; i < K_CLU; ++i) {
    float e = __expf(p[(size_t)i * K_TOK] - mx);
    p[(size_t)i * K_TOK] = e;
    s += e;
  }
  float inv = 1.f / s;
  for (int i = 0; i < K_CLU; ++i) p[(size_t)i * K_TOK] *= inv;
}

// row renorm a_c /= (sum_k + 1e-5), convert to bf16 A-operand
__global__ void crow_renorm_kernel(const float* __restrict__ Sc,
                                   u16_t* __restrict__ Ac) {
  int row = blockIdx.x;                                  // 8192 rows (bh*256+c)
  const float* src = Sc + (size_t)row * K_TOK;
  float s = 0.f;
  for (int i = threadIdx.x; i < K_TOK; i += 256) s += src[i];
  __shared__ float red[256];
  red[threadIdx.x] = s;
  __syncthreads();
  for (int st = 128; st > 0; st >>= 1) {
    if ((int)threadIdx.x < st) red[threadIdx.x] += red[threadIdx.x + st];
    __syncthreads();
  }
  float inv = 1.f / (red[0] + 1e-5f);
  u16_t* dst = Ac + (size_t)row * K_TOK;
  for (int i = threadIdx.x; i < K_TOK; i += 256) dst[i] = f32_to_bf16(src[i] * inv);
}

// clusters PV: AO[b][2048+c][h*32+hd] = a_c @ v
__global__ void cpv_kernel(const u16_t* __restrict__ Ac,
                           const u16_t* __restrict__ Vt,
                           u16_t* __restrict__ AO) {
  int wid = (blockIdx.x * blockDim.x + threadIdx.x) >> 5;
  if (wid >= K_B * K_HEADS * (K_CLU / 16) * (K_HD / 16)) return;
  int bh = wid >> 5;                                     // 32 tiles per bh
  int t  = wid & 31;
  int mt = t >> 1, nt = t & 1;
  const u16_t* A  = Ac + ((size_t)bh * K_CLU + mt * 16) * K_TOK;
  const u16_t* Bt = Vt + ((size_t)bh * K_HD + nt * 16) * K_TOK;
  v8f acc = {};
  for (int k = 0; k < K_TOK; k += 32) {
    if (k + 64 <= K_TOK) { __builtin_prefetch(A + k + 32, 0, 1); __builtin_prefetch(Bt + k + 32, 0, 1); }
    acc = wmma_bf16(load_frag_a(A + k, K_TOK), load_frag_b(Bt + k, K_TOK), acc);
  }
  const int lane = threadIdx.x & 31, half = lane >> 4;
  const int b = bh >> 4, h = bh & 15;
  const int d = h * K_HD + nt * 16 + (lane & 15);
  #pragma unroll
  for (int r = 0; r < 8; ++r) {
    int c = mt * 16 + r + 8 * half;
    AO[((size_t)b * K_NQ + K_TOK + c) * K_D + d] = f32_to_bf16(acc[r]);
  }
}

// ---------------------------------------------------------------------------
// Final projection: (AO @ Wo) -> d_out (f32), row-dependent output placement
// ---------------------------------------------------------------------------
__global__ void outproj_kernel(const u16_t* __restrict__ AO,
                               const u16_t* __restrict__ WoT,
                               float* __restrict__ out) {
  int wid = (blockIdx.x * blockDim.x + threadIdx.x) >> 5;
  const int TPB = (K_NQ / 16) * (K_D / 16);              // 144*32
  if (wid >= K_B * TPB) return;
  int b = wid / TPB, t = wid % TPB;
  int mt = t / (K_D / 16), nt = t % (K_D / 16);
  const u16_t* A  = AO + ((size_t)b * K_NQ + mt * 16) * K_D;
  const u16_t* Bt = WoT + (size_t)nt * 16 * K_D;
  v8f acc = {};
  for (int k = 0; k < K_D; k += 32) {
    if (k + 64 <= K_D) { __builtin_prefetch(A + k + 32, 0, 1); __builtin_prefetch(Bt + k + 32, 0, 1); }
    acc = wmma_bf16(load_frag_a(A + k, K_D), load_frag_b(Bt + k, K_D), acc);
  }
  const int lane = threadIdx.x & 31, half = lane >> 4;
  const int col = nt * 16 + (lane & 15);
  #pragma unroll
  for (int r = 0; r < 8; ++r) {
    int m = mt * 16 + r + 8 * half;
    float* dst = (m < K_TOK)
        ? out + ((size_t)b * K_TOK + m) * K_D                                  // x_out
        : out + (size_t)K_B * K_TOK * K_D + ((size_t)b * K_CLU + (m - K_TOK)) * K_D; // clusters_out
    dst[col] = acc[r];
  }
}

// ---------------------------------------------------------------------------
// Launch
// ---------------------------------------------------------------------------
extern "C" void kernel_launch(void* const* d_in, const int* in_sizes, int n_in,
                              void* d_out, int out_size, void* d_ws, size_t ws_size,
                              hipStream_t stream) {
  (void)in_sizes; (void)n_in; (void)out_size; (void)ws_size;
  const float* x        = (const float*)d_in[0];
  const float* clusters = (const float*)d_in[1];
  // d_in[2] = token_sizes: additive per-query constant under softmax -> no-op
  const float* Wq  = (const float*)d_in[3];
  const float* Wkv = (const float*)d_in[4];
  const float* Wo  = (const float*)d_in[5];
  float* out = (float*)d_out;

  char* ws = (char*)d_ws;
  size_t off = 0;
  auto take = [&](size_t bytes) { char* p = ws + off; off += (bytes + 255) & ~(size_t)255; return p; };
  u16_t* Xbf  = (u16_t*)take((size_t)K_B * K_NQ * K_D * 2);            // 4.7 MB
  u16_t* WqT  = (u16_t*)take((size_t)K_D * K_D * 2);                   // 0.5 MB
  u16_t* WkvT = (u16_t*)take((size_t)2 * K_D * K_D * 2);               // 1.0 MB
  u16_t* WoT  = (u16_t*)take((size_t)K_D * K_D * 2);                   // 0.5 MB
  u16_t* Q    = (u16_t*)take((size_t)K_B * K_HEADS * K_NQ * K_HD * 2); // 4.7 MB
  u16_t* Kbuf = (u16_t*)take((size_t)K_B * K_HEADS * K_TOK * K_HD * 2);// 4.2 MB
  u16_t* Vt   = (u16_t*)take((size_t)K_B * K_HEADS * K_HD * K_TOK * 2);// 4.2 MB
  u16_t* AO   = (u16_t*)take((size_t)K_B * K_NQ * K_D * 2);            // 4.7 MB
  u16_t* Ac   = (u16_t*)take((size_t)K_B * K_HEADS * K_CLU * K_TOK * 2);// 33.6 MB
  float* Sc   = (float*)take((size_t)K_B * K_HEADS * K_CLU * K_TOK * 4);// 67.1 MB

  // prep
  transpose_to_bf16<<<(K_D * K_D + 255) / 256, 256, 0, stream>>>(Wq, WqT, K_D, K_D);
  transpose_to_bf16<<<(K_D * 2 * K_D + 255) / 256, 256, 0, stream>>>(Wkv, WkvT, K_D, 2 * K_D);
  transpose_to_bf16<<<(K_D * K_D + 255) / 256, 256, 0, stream>>>(Wo, WoT, K_D, K_D);
  pack_inputs<<<(K_B * K_NQ * K_D + 255) / 256, 256, 0, stream>>>(x, clusters, Xbf);

  // projections
  qproj_kernel<<<(K_B * 144 * 32) / 8, 256, 0, stream>>>(Xbf, WqT, Q);        // 1152 blocks
  kvproj_kernel<<<(K_B * 128 * 64) / 8, 256, 0, stream>>>(Xbf, WkvT, Kbuf, Vt); // 2048 blocks

  // x branch (fused flash attention)
  flash_x_kernel<<<K_B * K_HEADS * (K_TOK / 64), 128, 0, stream>>>(Q, Kbuf, Vt, AO);

  // cluster branch
  cscore_kernel<<<(K_B * K_HEADS * 16 * 128) / 8, 256, 0, stream>>>(Q, Kbuf, Sc);
  csoftmax_col_kernel<<<(K_B * K_HEADS * K_TOK) / 256, 256, 0, stream>>>(Sc);
  crow_renorm_kernel<<<K_B * K_HEADS * K_CLU, 256, 0, stream>>>(Sc, Ac);
  cpv_kernel<<<(K_B * K_HEADS * 16 * 2) / 8, 256, 0, stream>>>(Ac, Vt, AO);

  // output projection for both branches
  outproj_kernel<<<(K_B * 144 * 32) / 8, 256, 0, stream>>>(AO, WoT, out);
}
